// SpikingMLP_67216238182974
// MI455X (gfx1250) — matrix operand order repaired and checked by reading, not compiled, not run on hardware
//
#include <hip/hip_runtime.h>
#include <hip/hip_fp16.h>

// ---------------------------------------------------------------------------
// SpikingMLP forward for MI455X (gfx1250, wave32, WMMA f32_16x16x32_f16)
//
// Pipeline (all on `stream`):
//   k_if_input : IF neuron on raw input -> f16 spikes s0 [T*B, 32768]
//   k_gemm     : C[M,N] = spikes(f16) x W(fp32->f16 in-reg) (+bias), WMMA
//   k_bn_stats : per-feature mean/var over T*B=512 -> scale/shift
//   k_bn_if    : BN affine + IF neuron -> next-layer f16 spikes
//   k_mean_out : mean over T + output bias -> [64,1000] f32
//
// Roofline: W0 (256 MB fp32) dominates traffic; with 256(M)x64(N) workgroup
// tiles it streams from HBM exactly M/256 = 2x (~22 us @ 23.3 TB/s) while the
// 32 MB f16 spike matrix stays L2-resident. fp32->f16 conversion is fused in
// the GEMM (no weight-conversion scratch pass). Output N=1000 padded to 1024;
// pad columns compute garbage that is never read (mean_out guards o<1000).
// ---------------------------------------------------------------------------

typedef __attribute__((ext_vector_type(8)))  _Float16 v8h;
typedef __attribute__((ext_vector_type(16))) _Float16 v16h;
typedef __attribute__((ext_vector_type(16))) float    v16f;
typedef __attribute__((ext_vector_type(8)))  float    v8f;

#define V_TH   1.0f
#define BN_EPS 1e-5f

// ---------------- IF neuron on raw input -> f16 spikes ---------------------
__global__ void k_if_input(const float* __restrict__ x, _Float16* __restrict__ s0,
                           int n_per_t, int T) {
  int idx = blockIdx.x * blockDim.x + threadIdx.x;
  if (idx >= n_per_t) return;
  float v = 0.f;
  for (int t = 0; t < T; ++t) {
    v += x[(size_t)t * n_per_t + idx];
    float s = (v >= V_TH) ? 1.f : 0.f;
    v *= (1.f - s);                       // hard reset to 0
    s0[(size_t)t * n_per_t + idx] = (_Float16)s;
  }
}

// ---------------- WMMA fragment loaders ------------------------------------
// A 16x32 f16 tile (row-major A[M,K]); lane r=lane&15 half=lane>>4:
//   elements 0..15 = K {k0 + 16h .. +15}
__device__ __forceinline__ v16h load_a_frag(const _Float16* __restrict__ A,
                                            int row0, int K, int k0, int lane) {
  const int r = lane & 15, h = lane >> 4;
  const _Float16* p = A + (size_t)(row0 + r) * K + k0 + h * 16;
  v8h lo = *(const v8h*)p;          // 16B aligned
  v8h hi = *(const v8h*)(p + 8);
  v16h a;
#pragma unroll
  for (int i = 0; i < 8; ++i) { a[i] = lo[i]; a[i + 8] = hi[i]; }
  return a;
}

// B 32x16 f16 tile from fp32 weights W[N,K] row-major; lane c=lane&15 h=lane>>4
// holds col = col0+c, K = {k0 + 16h .. +15} -> 16 contiguous fp32, cvt to f16.
// Branchless: pad columns are clamped to a valid row (their results are
// garbage but never consumed), keeping the hot loop free of exec-mask ops.
__device__ __forceinline__ v16h load_b_frag_f32(const float* __restrict__ Wrow,
                                                int k0, int h) {
  const float* p = Wrow + k0 + h * 16;                  // 64B aligned
  v16f f = *(const v16f*)p;
  __builtin_prefetch(p + 256, 0, 2);                    // stream weights ahead
  v16h b;
#pragma unroll
  for (int i = 0; i < 16; ++i) b[i] = (_Float16)f[i];
  return b;
}

// ---------------- GEMM: C[M,N] = A[M,K](f16) * W[N,K]^T(f32) + bias --------
// Block: 256 threads = 8 waves, tile 256(M) x 64(N).
// Wave (wm=wave>>1, wn=wave&1) computes 64(M) x 32(N) = 4x2 WMMA tiles.
__global__ void __launch_bounds__(256)
k_gemm(const _Float16* __restrict__ A, const float* __restrict__ W,
       const float* __restrict__ bias, float* __restrict__ C,
       int K, int Nw, int ldc) {
  const int lane = threadIdx.x & 31;
  const int wave = threadIdx.x >> 5;
  const int wm = wave >> 1;                 // 0..3
  const int wn = wave & 1;                  // 0..1
  const int m0 = blockIdx.y * 256 + wm * 64;
  const int n0 = blockIdx.x * 64 + wn * 32;
  const int cc = lane & 15;                 // col within 16-tile
  const int h  = lane >> 4;                 // half selector

  // Per-lane weight row pointers (clamped for pad columns; see note above).
  const float* Wrow[2];
#pragma unroll
  for (int ni = 0; ni < 2; ++ni) {
    int col = n0 + ni * 16 + cc;
    if (col >= Nw) col = Nw - 1;            // clamp: valid memory, unused result
    Wrow[ni] = W + (size_t)col * K;
  }

  v8f acc[4][2];
#pragma unroll
  for (int mi = 0; mi < 4; ++mi)
#pragma unroll
    for (int ni = 0; ni < 2; ++ni)
#pragma unroll
      for (int e = 0; e < 8; ++e) acc[mi][ni][e] = 0.f;

  for (int k0 = 0; k0 < K; k0 += 32) {
    v16h a[4];
#pragma unroll
    for (int mi = 0; mi < 4; ++mi)
      a[mi] = load_a_frag(A, m0 + mi * 16, K, k0, lane);
#pragma unroll
    for (int ni = 0; ni < 2; ++ni) {
      v16h b = load_b_frag_f32(Wrow[ni], k0, h);
#pragma unroll
      for (int mi = 0; mi < 4; ++mi) {
        acc[mi][ni] = __builtin_amdgcn_wmma_f32_16x16x32_f16(
            /*neg_a=*/false, a[mi], /*neg_b=*/false, b,
            /*c_mod=*/(short)0, acc[mi][ni],
            /*reuse_a=*/false, /*reuse_b=*/false);
      }
    }
  }

  // Epilogue: add bias (per column), store f32.  C is padded, no col guard.
#pragma unroll
  for (int ni = 0; ni < 2; ++ni) {
    const int col = n0 + ni * 16 + cc;
    float bv = (bias != nullptr && col < Nw) ? bias[col] : 0.f;
#pragma unroll
    for (int mi = 0; mi < 4; ++mi) {
      const int row = m0 + mi * 16 + h * 8;
#pragma unroll
      for (int e = 0; e < 8; ++e)
        C[(size_t)(row + e) * ldc + col] = acc[mi][ni][e] + bv;
    }
  }
}

// ---------------- BatchNorm statistics over M=T*B samples ------------------
__global__ void k_bn_stats(const float* __restrict__ pre,
                           const float* __restrict__ g, const float* __restrict__ be,
                           float* __restrict__ scale, float* __restrict__ shift,
                           int M, int N) {
  int o = blockIdx.x * blockDim.x + threadIdx.x;
  if (o >= N) return;
  float s = 0.f, sq = 0.f;
  for (int m = 0; m < M; ++m) {
    float v = pre[(size_t)m * N + o];
    s += v; sq += v * v;
  }
  float mean = s / (float)M;
  float var  = sq / (float)M - mean * mean;   // biased, matches torch train BN
  float sc   = g[o] * rsqrtf(var + BN_EPS);
  scale[o] = sc;
  shift[o] = be[o] - mean * sc;
}

// ---------------- BN affine + IF neuron -> f16 spikes ----------------------
__global__ void k_bn_if(const float* __restrict__ pre,
                        const float* __restrict__ scale, const float* __restrict__ shift,
                        _Float16* __restrict__ s, int n_per_t, int N, int T) {
  int idx = blockIdx.x * blockDim.x + threadIdx.x;
  if (idx >= n_per_t) return;
  int o = idx % N;
  float sc = scale[o], sh = shift[o];
  float v = 0.f;
  for (int t = 0; t < T; ++t) {
    float y = pre[(size_t)t * n_per_t + idx] * sc + sh;
    v += y;
    float sp = (v >= V_TH) ? 1.f : 0.f;
    v *= (1.f - sp);
    s[(size_t)t * n_per_t + idx] = (_Float16)sp;
  }
}

// ---------------- mean over T + output bias --------------------------------
__global__ void k_mean_out(const float* __restrict__ pre, const float* __restrict__ bout,
                           float* __restrict__ out, int B, int NC, int ldc, int T) {
  int idx = blockIdx.x * blockDim.x + threadIdx.x;
  if (idx >= B * NC) return;
  int b = idx / NC, o = idx % NC;
  float sum = 0.f;
  for (int t = 0; t < T; ++t)
    sum += pre[(size_t)(t * B + b) * ldc + o];
  out[idx] = sum * (1.f / (float)T) + bout[o];
}

// ---------------------------------------------------------------------------
extern "C" void kernel_launch(void* const* d_in, const int* in_sizes, int n_in,
                              void* d_out, int out_size, void* d_ws, size_t ws_size,
                              hipStream_t stream) {
  const float* x    = (const float*)d_in[0];
  const float* W0   = (const float*)d_in[1];
  const float* b0   = (const float*)d_in[2];
  const float* g0   = (const float*)d_in[3];
  const float* be0  = (const float*)d_in[4];
  const float* W1   = (const float*)d_in[5];
  const float* b1   = (const float*)d_in[6];
  const float* g1   = (const float*)d_in[7];
  const float* be1  = (const float*)d_in[8];
  const float* W2   = (const float*)d_in[9];
  const float* b2   = (const float*)d_in[10];
  const float* g2   = (const float*)d_in[11];
  const float* be2  = (const float*)d_in[12];
  const float* Wout = (const float*)d_in[13];
  const float* bout = (const float*)d_in[14];
  float* out = (float*)d_out;

  const int T = 8, B = 64, F0 = 32768, D = 2048, NC = 1000;
  const int M = T * B;            // 512
  const int NPAD = 1024;          // padded output cols (>= NC, mult of 64)

  // Workspace carve-out (~42 MB)
  char* ws = (char*)d_ws;
  size_t off = 0;
  _Float16* s0 = (_Float16*)(ws + off); off += (size_t)M * F0 * sizeof(_Float16); // 32 MB
  _Float16* sA = (_Float16*)(ws + off); off += (size_t)M * D * sizeof(_Float16);  // 2 MB
  _Float16* sB = (_Float16*)(ws + off); off += (size_t)M * D * sizeof(_Float16);  // 2 MB
  float* pre   = (float*)(ws + off);    off += (size_t)M * D * sizeof(float);     // 4 MB
  float* scale = (float*)(ws + off);    off += (size_t)D * sizeof(float);
  float* shift = (float*)(ws + off);    off += (size_t)D * sizeof(float);
  (void)ws_size; (void)in_sizes; (void)n_in; (void)out_size;

  const dim3 blk(256);

  // sn0: IF on raw input
  int npt0 = B * F0;
  k_if_input<<<dim3((npt0 + 255) / 256), blk, 0, stream>>>(x, s0, npt0, T);

  // Layer 0: [512,32768] x [2048,32768]^T
  k_gemm<<<dim3(D / 64, M / 256), blk, 0, stream>>>(s0, W0, b0, pre, F0, D, D);
  k_bn_stats<<<dim3(D / 256), blk, 0, stream>>>(pre, g0, be0, scale, shift, M, D);
  k_bn_if<<<dim3((B * D + 255) / 256), blk, 0, stream>>>(pre, scale, shift, sA, B * D, D, T);

  // Layer 1
  k_gemm<<<dim3(D / 64, M / 256), blk, 0, stream>>>(sA, W1, b1, pre, D, D, D);
  k_bn_stats<<<dim3(D / 256), blk, 0, stream>>>(pre, g1, be1, scale, shift, M, D);
  k_bn_if<<<dim3((B * D + 255) / 256), blk, 0, stream>>>(pre, scale, shift, sB, B * D, D, T);

  // Layer 2
  k_gemm<<<dim3(D / 64, M / 256), blk, 0, stream>>>(sB, W2, b2, pre, D, D, D);
  k_bn_stats<<<dim3(D / 256), blk, 0, stream>>>(pre, g2, be2, scale, shift, M, D);
  k_bn_if<<<dim3((B * D + 255) / 256), blk, 0, stream>>>(pre, scale, shift, sA, B * D, D, T);

  // Output layer: [512,2048] x [1000,2048]^T -> padded [512,1024]; bias in mean.
  k_gemm<<<dim3(NPAD / 64, M / 256), blk, 0, stream>>>(sA, Wout, nullptr, pre, D, NC, NPAD);
  k_mean_out<<<dim3((B * NC + 255) / 256), blk, 0, stream>>>(pre, bout, out, B, NC, NPAD, T);
}
